// MultiScaleBlock_50749333569666
// MI455X (gfx1250) — matrix operand order repaired and checked by reading, not compiled, not run on hardware
//
#include <hip/hip_runtime.h>
#include <hip/hip_bf16.h>
#include <stdint.h>

// ---------------------------------------------------------------------------
// CDNA5 (gfx1250) MultiScaleBlock forward.
// GEMMs: v_wmma_f32_16x16x32_bf16, bf16 operands staged into LDS with
// global_load_async_to_lds_b128 (ASYNCcnt), f32 accumulate.
// ---------------------------------------------------------------------------

typedef __attribute__((ext_vector_type(16))) __bf16          v16bf;
typedef __attribute__((ext_vector_type(16))) unsigned short  v16us;
typedef __attribute__((ext_vector_type(8)))  float           v8f;

#define GEMM_BM 128
#define GEMM_BN 128
#define GEMM_KT 32
#define GEMM_LDK 40   // padded LDS row stride (halfs): 80B rows, 16B aligned,
                      // 20*m mod 64 distinct for 16 consecutive rows

__device__ __forceinline__ unsigned short f2bf(float f) {
    unsigned int u = __float_as_uint(f);
    unsigned int r = (u + 0x7FFFu + ((u >> 16) & 1u)) >> 16;   // RNE
    return (unsigned short)r;
}
__device__ __forceinline__ float bf2f(unsigned short u) {
    return __uint_as_float(((unsigned int)u) << 16);
}
__device__ __forceinline__ float gelu_f(float x) {
    return 0.5f * x * (1.0f + erff(x * 0.70710678118654752f));
}

// 16-byte async copy global -> LDS (tracked by ASYNCcnt)
__device__ __forceinline__ void async_cp16(unsigned lds_off, const unsigned short* g) {
    asm volatile("global_load_async_to_lds_b128 %0, %1, off"
                 :: "v"(lds_off), "v"(g) : "memory");
}
__device__ __forceinline__ void async_wait0() {
    asm volatile("s_wait_asynccnt 0" ::: "memory");
}

// A-matrix fragment (16x32 bf16, MxK). ISA 7.12.2: lane l holds row m=l&15,
// lanes 0-15 cover K {0..7,16..23}, lanes 16-31 cover K {8..15,24..31}.
__device__ __forceinline__ v16bf frag_a(const unsigned short* t, int row0, int lane) {
    int r  = row0 + (lane & 15);
    int kb = (lane >> 4) << 3;
    const unsigned short* p = t + r * GEMM_LDK + kb;
    v16us u;
#pragma unroll
    for (int j = 0; j < 8; ++j) u[j] = p[j];
#pragma unroll
    for (int j = 0; j < 8; ++j) u[8 + j] = p[16 + j];
    return __builtin_bit_cast(v16bf, u);
}
// B-matrix fragment (32x16 bf16, KxN), tile stored [n][k] (== W memory layout).
__device__ __forceinline__ v16bf frag_b(const unsigned short* t, int row0, int lane) {
    int r  = row0 + (lane & 15);
    int kb = (lane >> 4) << 3;
    const unsigned short* p = t + r * GEMM_LDK + kb;
    v16us u;
#pragma unroll
    for (int j = 0; j < 8; ++j) u[j] = p[j];
#pragma unroll
    for (int j = 0; j < 8; ++j) u[8 + j] = p[16 + j];
    return __builtin_bit_cast(v16bf, u);
}

#define WMMA_BF16(a, b, c) \
    __builtin_amdgcn_wmma_f32_16x16x32_bf16(false, (a), false, (b), (short)0, (c), false, false)

// O[M,N] = A[M,K] @ W[N,K]^T + bias  (+GELU flags&1) (+R[M,N] flags&2)
// flags&4: write bf16 output, else f32. Requires N%128==0, K%32==0; M guarded.
__global__ __launch_bounds__(256) void gemm_kernel(
    const unsigned short* __restrict__ A, const unsigned short* __restrict__ W,
    const float* __restrict__ bias, const float* __restrict__ R,
    void* __restrict__ Optr, int M, int N, int K, int flags)
{
    __shared__ unsigned short As[GEMM_BM * GEMM_LDK];
    __shared__ unsigned short Ws[GEMM_BN * GEMM_LDK];

    const int tid  = threadIdx.x;
    const int lane = tid & 31;
    const int wid  = tid >> 5;
    const int wm   = wid >> 1;          // 0..3, 32 rows each
    const int wn   = wid & 1;           // 0..1, 64 cols each
    const int m0   = blockIdx.y * GEMM_BM;
    const int n0   = blockIdx.x * GEMM_BN;
    const bool interior = (m0 + GEMM_BM) <= M;

    const unsigned as_lds = (unsigned)(uintptr_t)(&As[0]);
    const unsigned ws_lds = (unsigned)(uintptr_t)(&Ws[0]);

    v8f acc[2][4];
#pragma unroll
    for (int i = 0; i < 2; ++i)
#pragma unroll
        for (int j = 0; j < 4; ++j) acc[i][j] = (v8f){};

    for (int kt = 0; kt < K; kt += GEMM_KT) {
        __syncthreads();
        // W tile: 128x32 halfs = 512 16B chunks, 2 per thread, async to LDS
#pragma unroll
        for (int c = tid; c < 512; c += 256) {
            int r = c >> 2, col = (c & 3) * 8;
            async_cp16(ws_lds + (unsigned)(r * GEMM_LDK + col) * 2,
                       W + (size_t)(n0 + r) * K + kt + col);
        }
        if (interior) {
#pragma unroll
            for (int c = tid; c < 512; c += 256) {
                int r = c >> 2, col = (c & 3) * 8;
                async_cp16(as_lds + (unsigned)(r * GEMM_LDK + col) * 2,
                           A + (size_t)(m0 + r) * K + kt + col);
            }
        } else {
#pragma unroll
            for (int c = tid; c < 512; c += 256) {
                int r = c >> 2, col = (c & 3) * 8;
                int gm = m0 + r;
                uint4 v = make_uint4(0u, 0u, 0u, 0u);
                if (gm < M) v = *(const uint4*)(A + (size_t)gm * K + kt + col);
                *(uint4*)(&As[r * GEMM_LDK + col]) = v;
            }
        }
        async_wait0();
        __syncthreads();

        v16bf a0 = frag_a(As, wm * 32,      lane);
        v16bf a1 = frag_a(As, wm * 32 + 16, lane);
        v16bf b0 = frag_b(Ws, wn * 64,      lane);
        v16bf b1 = frag_b(Ws, wn * 64 + 16, lane);
        v16bf b2 = frag_b(Ws, wn * 64 + 32, lane);
        v16bf b3 = frag_b(Ws, wn * 64 + 48, lane);

        acc[0][0] = WMMA_BF16(a0, b0, acc[0][0]);
        acc[0][1] = WMMA_BF16(a0, b1, acc[0][1]);
        acc[0][2] = WMMA_BF16(a0, b2, acc[0][2]);
        acc[0][3] = WMMA_BF16(a0, b3, acc[0][3]);
        acc[1][0] = WMMA_BF16(a1, b0, acc[1][0]);
        acc[1][1] = WMMA_BF16(a1, b1, acc[1][1]);
        acc[1][2] = WMMA_BF16(a1, b2, acc[1][2]);
        acc[1][3] = WMMA_BF16(a1, b3, acc[1][3]);
    }

    // D layout: lane = col n=l&15; rows m = (l>>4)*8 + r
    const int col   = lane & 15;
    const int rbase = (lane >> 4) << 3;
    const bool out_bf = (flags & 4) != 0;
#pragma unroll
    for (int mi = 0; mi < 2; ++mi) {
#pragma unroll
        for (int nj = 0; nj < 4; ++nj) {
            int n = n0 + wn * 64 + nj * 16 + col;
            float bv = bias[n];
#pragma unroll
            for (int r = 0; r < 8; ++r) {
                int m = m0 + wm * 32 + mi * 16 + rbase + r;
                if (m < M) {
                    float v = acc[mi][nj][r] + bv;
                    if (flags & 1) v = gelu_f(v);
                    if (flags & 2) v += R[(size_t)m * N + n];
                    if (out_bf) ((unsigned short*)Optr)[(size_t)m * N + n] = f2bf(v);
                    else        ((float*)Optr)[(size_t)m * N + n] = v;
                }
            }
        }
    }
}

// f32 -> bf16 flat convert (weights, once per launch)
__global__ void cvt_kernel(const float* __restrict__ src,
                           unsigned short* __restrict__ dst, long n)
{
    for (long i = (long)blockIdx.x * blockDim.x + threadIdx.x; i < n;
         i += (long)gridDim.x * blockDim.x)
        dst[i] = f2bf(src[i]);
}

// LayerNorm over rows of length C; f32 in, bf16 out (optional fused GELU)
__global__ __launch_bounds__(256) void ln_kernel(
    const float* __restrict__ X, unsigned short* __restrict__ O, int C,
    const float* __restrict__ g, const float* __restrict__ b, int do_gelu)
{
    __shared__ float red[256];
    const int tid = threadIdx.x;
    const size_t row = blockIdx.x;
    const float* x = X + row * C;
    float s = 0.0f, s2 = 0.0f;
    for (int i = tid; i < C; i += 256) { float v = x[i]; s += v; s2 += v * v; }
    red[tid] = s; __syncthreads();
    for (int o = 128; o > 0; o >>= 1) { if (tid < o) red[tid] += red[tid + o]; __syncthreads(); }
    float mu = red[0] / C; __syncthreads();
    red[tid] = s2; __syncthreads();
    for (int o = 128; o > 0; o >>= 1) { if (tid < o) red[tid] += red[tid + o]; __syncthreads(); }
    float var = red[0] / C - mu * mu;
    float rs  = rsqrtf(var + 1e-5f);
    unsigned short* o = O + row * C;
    for (int i = tid; i < C; i += 256) {
        float y = (x[i] - mu) * rs * g[i] + b[i];
        if (do_gelu) y = gelu_f(y);
        o[i] = f2bf(y);
    }
}

// Masked softmax attention, hd=64, one wave32 per (b,h,q); bf16 in/out, f32 math.
__global__ __launch_bounds__(256) void attn_kernel(
    const unsigned short* __restrict__ Q, const unsigned short* __restrict__ Kp,
    const unsigned short* __restrict__ V, const int* __restrict__ mask,
    unsigned short* __restrict__ O, int B, int H, int Nq, int Nk, float scale,
    long qb, long qn, long kb, long kn, long ob, long on)
{
    const int HD = 64;
    __shared__ float qs[8][64];
    __shared__ float sc[8][416];
    const int lane = threadIdx.x & 31;
    const int wid  = threadIdx.x >> 5;
    long w = (long)blockIdx.x * 8 + wid;
    const long total = (long)B * H * Nq;
    const bool valid = (w < total);
    if (!valid) w = 0;
    const int q = (int)(w % Nq);
    const int h = (int)((w / Nq) % H);
    const int b = (int)(w / ((long)Nq * H));

    const unsigned short* qp = Q + (long)b * qb + (long)q * qn + h * HD;
    qs[wid][lane]      = bf2f(qp[lane])      * scale;
    qs[wid][lane + 32] = bf2f(qp[lane + 32]) * scale;
    __syncthreads();

    for (int n = lane; n < Nk; n += 32) {
        const unsigned short* kp = Kp + (long)b * kb + (long)n * kn + h * HD;
        float s = 0.0f;
#pragma unroll 8
        for (int d = 0; d < HD; ++d) s += qs[wid][d] * bf2f(kp[d]);
        if (mask[(long)b * Nk + n] == 0) s = -1e30f;
        sc[wid][n] = s;
    }
    __syncthreads();

    float m = -1e30f;
    for (int n = lane; n < Nk; n += 32) m = fmaxf(m, sc[wid][n]);
    for (int o = 16; o > 0; o >>= 1) m = fmaxf(m, __shfl_xor(m, o, 32));
    float sum = 0.0f;
    for (int n = lane; n < Nk; n += 32) {
        float p = __expf(sc[wid][n] - m);
        sc[wid][n] = p;
        sum += p;
    }
    for (int o = 16; o > 0; o >>= 1) sum += __shfl_xor(sum, o, 32);
    float inv = 1.0f / sum;
    __syncthreads();

    float a0 = 0.0f, a1 = 0.0f;
    for (int n = 0; n < Nk; ++n) {
        const unsigned short* vp = V + (long)b * kb + (long)n * kn + h * HD;
        float p = sc[wid][n];
        a0 += p * bf2f(vp[lane]);
        a1 += p * bf2f(vp[lane + 32]);
    }
    if (valid) {
        unsigned short* op = O + (long)b * ob + (long)q * on + h * HD;
        op[lane]      = f2bf(a0 * inv);
        op[lane + 32] = f2bf(a1 * inv);
    }
}

// Strided row copy: dst[r*ds + c] = src[r*ss + c]
template <typename T>
__global__ void copy_rows_kernel(const T* __restrict__ src, T* __restrict__ dst,
                                 long rows, long rowlen, long ss, long ds)
{
    long total = rows * rowlen;
    for (long i = (long)blockIdx.x * blockDim.x + threadIdx.x; i < total;
         i += (long)gridDim.x * blockDim.x) {
        long r = i / rowlen, c = i % rowlen;
        dst[r * ds + c] = src[r * ss + c];
    }
}

// ---------------------------------------------------------------------------
// Host-side orchestration
// ---------------------------------------------------------------------------

static void gemm(hipStream_t s, const unsigned short* A, const unsigned short* W,
                 const float* bias, const float* R, void* O,
                 int M, int N, int K, int flags) {
    dim3 grid(N / GEMM_BN, (M + GEMM_BM - 1) / GEMM_BM);
    gemm_kernel<<<grid, 256, 0, s>>>(A, W, bias, R, O, M, N, K, flags);
}
static void lnorm(hipStream_t s, const float* X, unsigned short* O, int M, int C,
                  const float* g, const float* b, int do_gelu) {
    ln_kernel<<<M, 256, 0, s>>>(X, O, C, g, b, do_gelu);
}
static void attn(hipStream_t s, const unsigned short* Q, const unsigned short* K,
                 const unsigned short* V, const int* mask, unsigned short* O,
                 int B, int H, int Nq, int Nk, float scale,
                 long qb, long qn, long kb, long kn, long ob, long on) {
    long waves = (long)B * H * Nq;
    int blocks = (int)((waves + 7) / 8);
    attn_kernel<<<blocks, 256, 0, s>>>(Q, K, V, mask, O, B, H, Nq, Nk, scale,
                                       qb, qn, kb, kn, ob, on);
}
template <typename T>
static void crows(hipStream_t s, const T* src, T* dst,
                  long rows, long rowlen, long ss, long ds) {
    long total = rows * rowlen;
    int blocks = (int)((total + 255) / 256);
    if (blocks > 4096) blocks = 4096;
    copy_rows_kernel<T><<<blocks, 256, 0, s>>>(src, dst, rows, rowlen, ss, ds);
}

extern "C" void kernel_launch(void* const* d_in, const int* in_sizes, int n_in,
                              void* d_out, int out_size, void* d_ws, size_t ws_size,
                              hipStream_t stream)
{
    (void)in_sizes; (void)n_in; (void)out_size; (void)ws_size;
    const int B = 32;
    const size_t SZ0 = (size_t)B * 401 * 384;   // 4,927,488
    const size_t SZ1 = (size_t)B * 197 * 768;   // 4,841,472

    const int* mask0 = (const int*)d_in[2];
    const int* mask1 = (const int*)d_in[3];
    auto F = [&](int k) -> const float* { return (const float*)d_in[k]; };

    // ---- workspace carve ----
    float* wsf  = (float*)d_ws;
    float* X0   = wsf;              // f32 residual streams + small f32 bufs
    float* X1   = X0 + SZ0;
    float* TMP  = X1 + SZ1;
    float* Sf   = TMP + SZ0;
    float* SC_IN = Sf;  Sf += 32 * 768;
    float* CLS0  = Sf;  Sf += 32 * 768;
    float* CLS1  = Sf;  Sf += 32 * 384;
    float* XR    = Sf;  Sf += 32 * 768;
    float* X0B   = Sf;  Sf += 32 * 768;
    float* FO    = Sf;  Sf += 32 * 768;
    float* REV   = Sf;  Sf += 32 * 768;

    unsigned short* bh  = (unsigned short*)Sf;  // bf16 region (16B aligned)
    unsigned short* T   = bh;                   bh += SZ0;
    unsigned short* QKV = bh;                   bh += 14782464;
    unsigned short* ATT = bh;                   bh += SZ0;
    unsigned short* HID = bh;                   bh += 19709952;
    unsigned short* QB    = bh;  bh += 32 * 768;
    unsigned short* AO    = bh;  bh += 32 * 768;
    unsigned short* LN0   = bh;  bh += 32 * 768;
    unsigned short* SC_LN = bh;  bh += 32 * 768;
    unsigned short* LN2s  = bh;  bh += 32 * 768;
    unsigned short* SH    = bh;  bh += 32 * 3072;
    unsigned short* LN3   = bh;  bh += 32 * 768;
    unsigned short* wb    = bh;                 // bf16 weight pool (~40.1M halfs)
    unsigned short* KB = QKV;
    unsigned short* VB = QKV + 5000000;

    // lazy weight f32->bf16 conversion (deterministic order per launch)
    unsigned short* wbcur = wb;
    auto WB = [&](int idx, long n) -> const unsigned short* {
        unsigned short* dst = wbcur; wbcur += (size_t)n;
        int blocks = (int)((n + 255) / 256);
        if (blocks > 8192) blocks = 8192;
        cvt_kernel<<<blocks, 256, 0, stream>>>(F(idx), dst, n);
        return dst;
    };

    hipMemcpyAsync(X0, d_in[0], SZ0 * sizeof(float), hipMemcpyDeviceToDevice, stream);
    hipMemcpyAsync(X1, d_in[1], SZ1 * sizeof(float), hipMemcpyDeviceToDevice, stream);

    // ---- transformer block: x += attn(ln1(x)); x += mlp(ln2(x)) ----
    auto run_block = [&](float* X, const int* mask, int Nt, int C, int H, int p) {
        int M = B * Nt;
        lnorm(stream, X, T, M, C, F(p + 0), F(p + 1), 0);
        gemm(stream, T, WB(p + 2, (long)3 * C * C), F(p + 3), nullptr, QKV, M, 3 * C, C, 4);
        attn(stream, QKV, QKV + C, QKV + 2 * C, mask, ATT, B, H, Nt, Nt, 0.125f,
             (long)Nt * 3 * C, 3L * C, (long)Nt * 3 * C, 3L * C,
             (long)Nt * C, (long)C);
        gemm(stream, ATT, WB(p + 4, (long)C * C), F(p + 5), X, X, M, C, C, 2);
        lnorm(stream, X, T, M, C, F(p + 6), F(p + 7), 0);
        gemm(stream, T, WB(p + 8, (long)4 * C * C), F(p + 9), nullptr, HID, M, 4 * C, C, 1 | 4);
        gemm(stream, HID, WB(p + 10, (long)C * 4 * C), F(p + 11), X, X, M, C, 4 * C, 2);
    };

    run_block(X0, mask0, 401, 384, 6, 4);
    for (int i = 0; i < 4; ++i) run_block(X1, mask1, 197, 768, 12, 16 + 12 * i);

    // ---- cls projections: proj_fwd(outs_b[i][:, :1]) ----
    crows<float>(stream, X0, SC_IN, B, 384, 401L * 384, 384);
    lnorm(stream, SC_IN, SC_LN, B, 384, F(64), F(65), 1);
    gemm(stream, SC_LN, WB(66, 768L * 384), F(67), nullptr, CLS0, B, 768, 384, 0);
    crows<float>(stream, X1, SC_IN, B, 768, 197L * 768, 768);
    lnorm(stream, SC_IN, SC_LN, B, 768, F(68), F(69), 1);
    gemm(stream, SC_LN, WB(70, 384L * 768), F(71), nullptr, CLS1, B, 384, 768, 0);

    // ---- fusion: ca_block on [cls_i ; outs_b[j][:,1:]], revert, assemble ----
    auto fuse = [&](int pf, int pv, const float* cls, const float* Xj, const int* maskj,
                    int Ck, int Nt, int H, const float* Xi, int Ci, int Ni, float* outp) {
        int M = B * Nt;
        crows<float>(stream, cls, TMP, B, Ck, Ck, (long)Nt * Ck);
        crows<float>(stream, Xj + Ck, TMP + Ck, B, (long)(Nt - 1) * Ck,
                     (long)Nt * Ck, (long)Nt * Ck);
        lnorm(stream, TMP, T, M, Ck, F(pf + 0), F(pf + 1), 0);
        crows<unsigned short>(stream, T, LN0, B, Ck, (long)Nt * Ck, Ck);
        gemm(stream, LN0, WB(pf + 2, (long)Ck * Ck), F(pf + 3), nullptr, QB, B, Ck, Ck, 4);
        gemm(stream, T, WB(pf + 4, (long)Ck * Ck), F(pf + 5), nullptr, KB, M, Ck, Ck, 4);
        gemm(stream, T, WB(pf + 6, (long)Ck * Ck), F(pf + 7), nullptr, VB, M, Ck, Ck, 4);
        attn(stream, QB, KB, VB, maskj, AO, B, H, 1, Nt, 0.125f,
             (long)Ck, 0L, (long)Nt * Ck, (long)Ck, (long)Ck, 0L);
        crows<float>(stream, TMP, XR, B, Ck, (long)Nt * Ck, Ck);
        gemm(stream, AO, WB(pf + 8, (long)Ck * Ck), F(pf + 9), XR, X0B, B, Ck, Ck, 2);
        lnorm(stream, X0B, LN2s, B, Ck, F(pf + 10), F(pf + 11), 0);
        gemm(stream, LN2s, WB(pf + 12, (long)4 * Ck * Ck), F(pf + 13), nullptr, SH, B, 4 * Ck, Ck, 1 | 4);
        gemm(stream, SH, WB(pf + 14, (long)Ck * 4 * Ck), F(pf + 15), X0B, FO, B, Ck, 4 * Ck, 2);
        lnorm(stream, FO, LN3, B, Ck, F(pv + 0), F(pv + 1), 1);
        gemm(stream, LN3, WB(pv + 2, (long)Ci * Ck), F(pv + 3), nullptr, REV, B, Ci, Ck, 0);
        crows<float>(stream, REV, outp, B, Ci, Ci, (long)Ni * Ci);
        crows<float>(stream, Xi + Ci, outp + Ci, B, (long)(Ni - 1) * Ci,
                     (long)Ni * Ci, (long)Ni * Ci);
    };

    float* out = (float*)d_out;
    fuse(72, 104, CLS0, X1, mask1, 768, 197, 12, X0, 384, 401, out);
    fuse(88, 108, CLS1, X0, mask0, 384, 401, 6,  X1, 768, 197, out + SZ0);
}
